// BiLSTM_CRF_28406913696562
// MI455X (gfx1250) — compile-verified, matrix operand order checked by reading, not run on hardware
//
#include <hip/hip_runtime.h>
#include <hip/hip_bf16.h>

#define B_ 64
#define T_ 512
#define E_ 512
#define H_ 512
#define NT_ 24
#define NTP_ 32
#define START_ 22
#define END_ 23

typedef __bf16 bf16_t;
typedef bf16_t v8bf  __attribute__((ext_vector_type(8)));
typedef bf16_t v16bf __attribute__((ext_vector_type(16)));
typedef float  v8f   __attribute__((ext_vector_type(8)));
typedef unsigned int v4u __attribute__((ext_vector_type(4)));
typedef int v8i __attribute__((ext_vector_type(8)));
typedef int v4i __attribute__((ext_vector_type(4)));

#if __has_builtin(__builtin_amdgcn_tensor_load_to_lds) && __has_builtin(__builtin_amdgcn_s_wait_tensorcnt)
#define HAVE_TDM 1
#endif

// Load a 16-element bf16 WMMA fragment (A or B) for this lane.
// CDNA5 16-bit A/B layout: lanes 0-15 hold K=[kb..kb+7] and [kb+16..kb+23],
// lanes 16-31 hold K=[kb+8..kb+15] and [kb+24..kb+31].
// caller passes off0 = kb + 8*(lane>>4); second run is off0+16.
__device__ __forceinline__ v16bf load_frag(const bf16_t* p, int off0) {
    v8bf lo = *(const v8bf*)(p + off0);
    v8bf hi = *(const v8bf*)(p + off0 + 16);
    return __builtin_shufflevector(lo, hi, 0,1,2,3,4,5,6,7,8,9,10,11,12,13,14,15);
}

__device__ __forceinline__ v16bf zero_frag() {
    v16bf z;
#pragma unroll
    for (int i = 0; i < 16; ++i) z[i] = (bf16_t)0.0f;
    return z;
}

__device__ __forceinline__ float sigf(float x) { return 1.0f / (1.0f + __expf(-x)); }

// TDM: DMA a [64 rows x 128 K] bf16 tile (row stride 512 elems) from global into LDS.
__device__ __forceinline__ void tdm_load_tile(const bf16_t* gsrc, bf16_t* lds_dst) {
#ifdef HAVE_TDM
    unsigned lds = (unsigned)(size_t)lds_dst;              // low 32 bits = LDS byte offset
    unsigned long long ga = (unsigned long long)(size_t)gsrc;
    v4u g0;
    g0[0] = 1u;                                            // count=1, user descriptor
    g0[1] = lds;                                           // lds_addr [63:32]
    g0[2] = (unsigned)(ga & 0xFFFFFFFFu);                  // global_addr [95:64]
    g0[3] = (unsigned)((ga >> 32) & 0x1FFFFFFu) | 0x80000000u; // addr[120:96] | type=2
    v8i g1;
    g1[0] = 0x00010000;            // workgroup_mask=0, data_size=1 (2 bytes)
    g1[1] = (int)(512u << 16);     // tensor_dim0=512 (bits[79:48], low part)
    g1[2] = (int)(2048u << 16);    // tensor_dim0 hi=0 ; tensor_dim1=2048 (low 16)
    g1[3] = (int)(128u << 16);     // tensor_dim1 hi=0 ; tile_dim0=128
    g1[4] = 64;                    // tile_dim1=64 ; tile_dim2=0
    g1[5] = 512;                   // tensor_dim0_stride=512 (low 32)
    g1[6] = 0;
    g1[7] = 0;
    v4i z4 = {0, 0, 0, 0};
#if __clang_major__ >= 23
    v8i z8 = {0, 0, 0, 0, 0, 0, 0, 0};
    __builtin_amdgcn_tensor_load_to_lds(g0, g1, z4, z4, z8, 0);
#else
    __builtin_amdgcn_tensor_load_to_lds(g0, g1, z4, z4, 0);
#endif
#endif
}

// ---------------------------------------------------------------- casts
__global__ void cast_f32_bf16(const float* __restrict__ s, bf16_t* __restrict__ d, long n) {
    long i = (long)blockIdx.x * blockDim.x + threadIdx.x;
    long stride = (long)gridDim.x * blockDim.x;
    for (; i < n; i += stride) d[i] = (bf16_t)s[i];
}

// pad W_out [24,1024] -> bf16 [32,1024] (rows >=24 are zero)
__global__ void pad_wout(const float* __restrict__ s, bf16_t* __restrict__ d) {
    int i = blockIdx.x * blockDim.x + threadIdx.x;   // 32*1024 threads
    if (i >= NTP_ * 2 * H_) return;
    int row = i / (2 * H_);
    d[i] = (row < NT_) ? (bf16_t)s[i] : (bf16_t)0.0f;
}

// ---------------------------------------------------------------- recurrent step GEMM
// G[dir][b][0:2048] = h[dir][b] @ W_hh[dir]^T + x_t[dir][b] @ W_ih[dir]^T
// grid: 64 blocks (32 ngroups x 2 dirs) x 128 threads (4 waves = 4 mtiles).
// Weight tiles (shared by the 4 waves) are staged in LDS via the Tensor Data Mover.
__global__ void __launch_bounds__(128)
step_gemm(const bf16_t* __restrict__ Xbf,   // [B,T,E] bf16
          const int*    __restrict__ len,   // [B]
          const bf16_t* __restrict__ Whh,   // [2][2048][512] bf16
          const bf16_t* __restrict__ Wih,   // [2][2048][512] bf16
          const bf16_t* __restrict__ Hb,    // [2][64][512] bf16
          float*        __restrict__ G,     // [2][64][2048] f32
          int t) {
    __shared__ bf16_t Lhh[64][128];   // 16 KB
    __shared__ bf16_t Lih[64][128];   // 16 KB

    const int lane = threadIdx.x & 31;
    const int ll = lane & 15, lh = lane >> 4;
    const int wv = threadIdx.x >> 5;           // mtile 0..3
    const int ng  = blockIdx.x & 31;
    const int dir = blockIdx.x >> 5;

    const int b = wv * 16 + ll;                // batch row this lane loads for A
    const bf16_t* Ah = Hb + ((size_t)(dir * B_ + b)) * H_;
    int te = t;
    if (dir) { int q = len[b] - 1 - t; te = q < 0 ? 0 : (q > T_ - 1 ? T_ - 1 : q); }
    const bf16_t* Ax = Xbf + ((size_t)b * T_ + te) * E_;

    const bf16_t* WhD = Whh + (size_t)dir * (4 * H_) * H_ + (size_t)(ng * 64) * H_;
    const bf16_t* WiD = Wih + (size_t)dir * (4 * H_) * E_ + (size_t)(ng * 64) * E_;

    v8f acc[4];
#pragma unroll
    for (int j = 0; j < 4; ++j)
#pragma unroll
        for (int r = 0; r < 8; ++r) acc[j][r] = 0.0f;

    for (int c = 0; c < 4; ++c) {              // K chunks of 128
        const int kbase = c * 128;
        // ---- stage weight tiles into LDS ----
#ifdef HAVE_TDM
        if (wv == 0) {
            tdm_load_tile(WhD + kbase, &Lhh[0][0]);
            tdm_load_tile(WiD + kbase, &Lih[0][0]);
            __builtin_amdgcn_s_wait_tensorcnt(0);
        }
#else
        for (int i = threadIdx.x; i < 1024; i += 128) {
            int row = i >> 4;
            int kl  = (i & 15) << 3;
            *(v8bf*)&Lhh[row][kl] = *(const v8bf*)(WhD + (size_t)row * H_ + kbase + kl);
            *(v8bf*)&Lih[row][kl] = *(const v8bf*)(WiD + (size_t)row * E_ + kbase + kl);
        }
#endif
        __syncthreads();

        // ---- compute: 4 K-steps of 32, 8 WMMAs each ----
#pragma unroll
        for (int kl = 0; kl < 128; kl += 32) {
            int off  = kl + (lh << 3);
            v16bf ah = load_frag(Ah, kbase + off);
            v16bf ax = load_frag(Ax, kbase + off);
            v16bf bh0 = load_frag(&Lhh[ll +  0][0], off);
            v16bf bh1 = load_frag(&Lhh[ll + 16][0], off);
            v16bf bh2 = load_frag(&Lhh[ll + 32][0], off);
            v16bf bh3 = load_frag(&Lhh[ll + 48][0], off);
            v16bf bi0 = load_frag(&Lih[ll +  0][0], off);
            v16bf bi1 = load_frag(&Lih[ll + 16][0], off);
            v16bf bi2 = load_frag(&Lih[ll + 32][0], off);
            v16bf bi3 = load_frag(&Lih[ll + 48][0], off);
            acc[0] = __builtin_amdgcn_wmma_f32_16x16x32_bf16(false, ah, false, bh0, (short)0, acc[0], false, false);
            acc[1] = __builtin_amdgcn_wmma_f32_16x16x32_bf16(false, ah, false, bh1, (short)0, acc[1], false, false);
            acc[2] = __builtin_amdgcn_wmma_f32_16x16x32_bf16(false, ah, false, bh2, (short)0, acc[2], false, false);
            acc[3] = __builtin_amdgcn_wmma_f32_16x16x32_bf16(false, ah, false, bh3, (short)0, acc[3], false, false);
            acc[0] = __builtin_amdgcn_wmma_f32_16x16x32_bf16(false, ax, false, bi0, (short)0, acc[0], false, false);
            acc[1] = __builtin_amdgcn_wmma_f32_16x16x32_bf16(false, ax, false, bi1, (short)0, acc[1], false, false);
            acc[2] = __builtin_amdgcn_wmma_f32_16x16x32_bf16(false, ax, false, bi2, (short)0, acc[2], false, false);
            acc[3] = __builtin_amdgcn_wmma_f32_16x16x32_bf16(false, ax, false, bi3, (short)0, acc[3], false, false);
        }
        __syncthreads();
    }

    // C/D layout: VGPR r -> row = wv*16 + r + 8*lh ; col = ng*64 + j*16 + ll
#pragma unroll
    for (int j = 0; j < 4; ++j) {
        int col = ng * 64 + j * 16 + ll;
#pragma unroll
        for (int r = 0; r < 8; ++r) {
            int bs = wv * 16 + r + (lh << 3);
            G[((size_t)(dir * B_ + bs)) * (4 * H_) + col] = acc[j][r];
        }
    }
}

// ---------------------------------------------------------------- gate / state update
// 2*64*512 = 65536 threads
__global__ void gate_update(const float* __restrict__ G,
                            const float* __restrict__ bias_f,
                            const float* __restrict__ bias_b,
                            const float* __restrict__ masks,   // [B,T]
                            bf16_t* __restrict__ Hb,           // [2][64][512]
                            float*  __restrict__ Cb,           // [2][64][512]
                            bf16_t* __restrict__ OutF,         // [B,T,H]
                            bf16_t* __restrict__ OutR,         // [B,T,H] (rev time order)
                            int t) {
    int idx = blockIdx.x * blockDim.x + threadIdx.x;
    if (idx >= 2 * B_ * H_) return;
    int dir = idx >> 15;
    int rem = idx & 32767;
    int b = rem >> 9;
    int u = rem & 511;

    const float* bias = dir ? bias_b : bias_f;
    size_t gbase = ((size_t)(dir * B_ + b)) * (4 * H_);
    float gi = G[gbase + 0 * H_ + u] + bias[0 * H_ + u];
    float gf = G[gbase + 1 * H_ + u] + bias[1 * H_ + u];
    float gg = G[gbase + 2 * H_ + u] + bias[2 * H_ + u];
    float go = G[gbase + 3 * H_ + u] + bias[3 * H_ + u];

    size_t sidx = ((size_t)(dir * B_ + b)) * H_ + u;
    float c_old = Cb[sidx];
    float h_old = (float)Hb[sidx];

    float i_g = sigf(gi), f_g = sigf(gf), o_g = sigf(go);
    float c_new = f_g * c_old + i_g * tanhf(gg);
    float h_new = o_g * tanhf(c_new);

    float m = masks[(size_t)b * T_ + t];
    float h_upd = m * h_new + (1.0f - m) * h_old;
    float c_upd = m * c_new + (1.0f - m) * c_old;

    Hb[sidx] = (bf16_t)h_upd;
    Cb[sidx] = c_upd;
    bf16_t* out = dir ? OutR : OutF;
    out[((size_t)b * T_ + t) * H_ + u] = (bf16_t)(h_upd * m);
}

// ---------------------------------------------------------------- emission GEMM
// emis[b,t,0:32] = [out_f | gathered out_rev] @ W_out^T + b_out
// 2048 mtile-waves, each handles both 16-col ntiles. 256 blocks x 256 threads.
__global__ void emis_gemm(const bf16_t* __restrict__ OutF,
                          const bf16_t* __restrict__ OutR,
                          const bf16_t* __restrict__ Wout,   // [32][1024] bf16
                          const float*  __restrict__ bout,   // [24]
                          const float*  __restrict__ masks,  // [B,T]
                          const int*    __restrict__ len,    // [B]
                          float*        __restrict__ Emis) { // [B*T][32]
    const int lane = threadIdx.x & 31;
    const int ll = lane & 15, lh = lane >> 4;
    const int mtile = (blockIdx.x << 3) | (threadIdx.x >> 5);

    const int rowA = mtile * 16 + ll;       // row in [0, B*T)
    const int b = rowA >> 9, t = rowA & 511;
    const float mval = masks[rowA];
    int q = len[b] - 1 - t;
    int ridx = q < 0 ? 0 : (q > T_ - 1 ? T_ - 1 : q);

    const bf16_t* Af = OutF + (size_t)rowA * H_;
    const bf16_t* Ab = OutR + ((size_t)b * T_ + ridx) * H_;
    const bf16_t* W0 = Wout + (size_t)(ll) * (2 * H_);
    const bf16_t* W1 = Wout + (size_t)(16 + ll) * (2 * H_);

    v8f acc0, acc1;
#pragma unroll
    for (int r = 0; r < 8; ++r) { acc0[r] = 0.0f; acc1[r] = 0.0f; }

#pragma unroll 2
    for (int kb = 0; kb < 2 * H_; kb += 32) {
        int off = kb + (lh << 3);
        v16bf a;
        if (kb < H_) {
            a = load_frag(Af, off);
        } else if (mval > 0.0f) {
            a = load_frag(Ab, off - H_);
        } else {
            a = zero_frag();
        }
        v16bf b0 = load_frag(W0, off);
        v16bf b1 = load_frag(W1, off);
        acc0 = __builtin_amdgcn_wmma_f32_16x16x32_bf16(false, a, false, b0, (short)0, acc0, false, false);
        acc1 = __builtin_amdgcn_wmma_f32_16x16x32_bf16(false, a, false, b1, (short)0, acc1, false, false);
    }

    int n0 = ll, n1 = 16 + ll;
    float bias0 = bout[n0];
    float bias1 = (n1 < NT_) ? bout[n1] : 0.0f;
#pragma unroll
    for (int r = 0; r < 8; ++r) {
        int mrow = mtile * 16 + r + (lh << 3);
        Emis[(size_t)mrow * NTP_ + n0] = acc0[r] + bias0;
        Emis[(size_t)mrow * NTP_ + n1] = acc1[r] + bias1;
    }
}

// ---------------------------------------------------------------- CRF Viterbi
// one wave (32 lanes) per batch element; lane = tag
__global__ void crf_viterbi(const float* __restrict__ Emis,   // [B*T][32]
                            const float* __restrict__ masks,  // [B,T]
                            const int*   __restrict__ len,
                            const float* __restrict__ trans,  // [24][24]
                            float* __restrict__ best_out,     // [B]
                            int*   __restrict__ seq_out,      // [B,T]
                            int*   __restrict__ BP) {         // [T][B][24]
    __shared__ float trs[NT_ * NT_];
    __shared__ float alpha_s[32];
    __shared__ float red[32];

    const int lane = threadIdx.x;
    const int b = blockIdx.x;

    for (int i = lane; i < NT_ * NT_; i += 32) trs[i] = trans[i];
    __syncthreads();

    float alpha = -1e30f;
    if (lane < NT_)
        alpha = Emis[((size_t)b * T_ + 0) * NTP_ + lane] + trs[START_ * NT_ + lane];

    for (int t = 1; t < T_; ++t) {
        alpha_s[lane] = alpha;
        __syncthreads();
        float m = masks[(size_t)b * T_ + t];
        float e = (lane < NT_) ? Emis[((size_t)b * T_ + t) * NTP_ + lane] : 0.0f;
        float best = -3.4e38f; int arg = 0;
#pragma unroll 4
        for (int p = 0; p < NT_; ++p) {
            float s = alpha_s[p] + trs[p * NT_ + lane];
            if (s > best) { best = s; arg = p; }
        }
        if (lane < NT_) BP[((size_t)t * B_ + b) * NT_ + lane] = arg;
        float mx = best + e;
        if (m > 0.0f) alpha = mx;
        __syncthreads();
    }

    float fin = (lane < NT_) ? alpha + trs[lane * NT_ + END_] : -3.4e38f;
    red[lane] = fin;
    __syncthreads();
    if (lane == 0) {
        float best = red[0]; int cur = 0;
        for (int p = 1; p < NT_; ++p)
            if (red[p] > best) { best = red[p]; cur = p; }
        best_out[b] = best;
        int last = len[b] - 1;
        for (int t = T_ - 1; t >= 0; --t) {
            seq_out[(size_t)b * T_ + t] = cur;
            if (t >= 1 && t <= last)
                cur = BP[((size_t)t * B_ + b) * NT_ + cur];
        }
    }
}

// ---------------------------------------------------------------- host
extern "C" void kernel_launch(void* const* d_in, const int* in_sizes, int n_in,
                              void* d_out, int out_size, void* d_ws, size_t ws_size,
                              hipStream_t stream) {
    const float* X      = (const float*)d_in[0];
    const float* masks  = (const float*)d_in[1];
    const int*   len    = (const int*)  d_in[2];
    const float* Wihf   = (const float*)d_in[3];
    const float* Whhf   = (const float*)d_in[4];
    const float* bf     = (const float*)d_in[5];
    const float* Wihb   = (const float*)d_in[6];
    const float* Whhb   = (const float*)d_in[7];
    const float* bb     = (const float*)d_in[8];
    const float* Wout   = (const float*)d_in[9];
    const float* bout   = (const float*)d_in[10];
    const float* trans  = (const float*)d_in[11];

    char* w = (char*)d_ws;
    bf16_t* Xbf   = (bf16_t*)w;  w += (size_t)B_ * T_ * E_ * 2;          // 33.5 MB
    bf16_t* Whh   = (bf16_t*)w;  w += (size_t)2 * 4 * H_ * H_ * 2;       // 4.2 MB
    bf16_t* Wih   = (bf16_t*)w;  w += (size_t)2 * 4 * H_ * E_ * 2;       // 4.2 MB
    bf16_t* Woutb = (bf16_t*)w;  w += (size_t)NTP_ * 2 * H_ * 2;         // 64 KB
    bf16_t* Hb    = (bf16_t*)w;  w += (size_t)2 * B_ * H_ * 2;           // 128 KB
    float*  Cb    = (float*)w;   w += (size_t)2 * B_ * H_ * 4;           // 256 KB
    float*  G     = (float*)w;   w += (size_t)2 * B_ * 4 * H_ * 4;       // 1 MB
    bf16_t* OutF  = (bf16_t*)w;  w += (size_t)B_ * T_ * H_ * 2;          // 33.5 MB
    bf16_t* OutR  = (bf16_t*)w;  w += (size_t)B_ * T_ * H_ * 2;          // 33.5 MB
    float*  Emis  = (float*)w;   w += (size_t)B_ * T_ * NTP_ * 4;        // 4.2 MB
    int*    BP    = (int*)w;     w += (size_t)T_ * B_ * NT_ * 4;         // 3.1 MB

    // casts to bf16
    {
        long n = (long)B_ * T_ * E_;
        cast_f32_bf16<<<2048, 256, 0, stream>>>(X, Xbf, n);
        long nw = (long)4 * H_ * H_;
        cast_f32_bf16<<<1024, 256, 0, stream>>>(Whhf, Whh, nw);
        cast_f32_bf16<<<1024, 256, 0, stream>>>(Whhb, Whh + nw, nw);
        cast_f32_bf16<<<1024, 256, 0, stream>>>(Wihf, Wih, nw);
        cast_f32_bf16<<<1024, 256, 0, stream>>>(Wihb, Wih + nw, nw);
        pad_wout<<<(NTP_ * 2 * H_ + 255) / 256, 256, 0, stream>>>(Wout, Woutb);
    }

    hipMemsetAsync(Hb, 0, (size_t)2 * B_ * H_ * 2, stream);
    hipMemsetAsync(Cb, 0, (size_t)2 * B_ * H_ * 4, stream);

    // recurrence: 512 steps, both directions per launch
    for (int t = 0; t < T_; ++t) {
        step_gemm<<<64, 128, 0, stream>>>(Xbf, len, Whh, Wih, Hb, G, t);
        gate_update<<<(2 * B_ * H_) / 256, 256, 0, stream>>>(G, bf, bb, masks, Hb, Cb, OutF, OutR, t);
    }

    // emissions
    emis_gemm<<<256, 256, 0, stream>>>(OutF, OutR, Woutb, bout, masks, len, Emis);

    // CRF decode
    float* best_out = (float*)d_out;
    int*   seq_out  = (int*)d_out + B_;
    crf_viterbi<<<B_, 32, 0, stream>>>(Emis, masks, len, trans, best_out, seq_out, BP);
}
